// GATBlock_23270132810342
// MI455X (gfx1250) — matrix operand order repaired and checked by reading, not compiled, not run on hardware
//
#include <hip/hip_runtime.h>

#define Bb  2
#define Nn  4096
#define Dd  128
#define Hh  4
#define DHh 32
#define NW  (Nn / 32)   // adj bitmask words per row

typedef __bf16 bf16_t;
typedef __attribute__((ext_vector_type(16))) __bf16 v16bf;
typedef __attribute__((ext_vector_type(8)))  __bf16 v8bf;
typedef __attribute__((ext_vector_type(8)))  float  v8f;

static __device__ __forceinline__ v16bf cat16(v8bf lo, v8bf hi) {
    return __builtin_shufflevector(lo, hi, 0, 1, 2, 3, 4, 5, 6, 7,
                                           8, 9, 10, 11, 12, 13, 14, 15);
}

// ---------------- Prep: fp32 -> bf16 convert of x -------------------------------
__global__ void cvt_x_kernel(const float* __restrict__ x, bf16_t* __restrict__ xb, int n) {
    int i = blockIdx.x * blockDim.x + threadIdx.x;
    if (i < n) xb[i] = (bf16_t)x[i];
}

// ---------------- Prep: transpose + convert weights (WT[sel][n][k] = W[k][n]) ----
__global__ void prep_w_kernel(const float* __restrict__ Wq, const float* __restrict__ Wk,
                              const float* __restrict__ Wv, const float* __restrict__ Wo,
                              bf16_t* __restrict__ WT) {
    const int sel = blockIdx.z;
    const float* W = (sel == 0) ? Wq : (sel == 1) ? Wk : (sel == 2) ? Wv : Wo;
    int i = blockIdx.x * blockDim.x + threadIdx.x;  // over D*D
    int k = i / Dd, n = i % Dd;
    WT[(size_t)sel * Dd * Dd + (size_t)n * Dd + k] = (bf16_t)W[(size_t)k * Dd + n];
}

// ---------------- Prep: pack adjacency into 1-bit mask (L2-resident, 2 MB) -------
__global__ void pack_adj_kernel(const int* __restrict__ adj, unsigned* __restrict__ bits) {
    int wid  = blockIdx.x * (blockDim.x >> 5) + (threadIdx.x >> 5);  // word index
    int lane = threadIdx.x & 31;
    int row  = wid >> 7;          // / NW
    int w    = wid & (NW - 1);    // % NW
    int v    = adj[(size_t)row * Nn + (w << 5) + lane];
    unsigned mask = __builtin_amdgcn_ballot_w32(v != 0);
    if (lane == 0) bits[wid] = mask;
}

// ---------------- Kernel 1: fused QKV projection (bf16 WMMA, f32 accum) ----------
__global__ void qkv_proj_kernel(const bf16_t* __restrict__ xb,
                                const bf16_t* __restrict__ WT,
                                const float* __restrict__ bq, const float* __restrict__ bk,
                                const float* __restrict__ bv,
                                bf16_t* __restrict__ q_bf, bf16_t* __restrict__ k_bf,
                                bf16_t* __restrict__ vT_bf) {
    const int lane = threadIdx.x & 31;
    const int r0   = blockIdx.x * 16;   // rows over flattened B*N
    const int c0   = blockIdx.y * 16;   // cols over D
    const int sel  = blockIdx.z;        // 0=q 1=k 2=v
    const float* bias = (sel == 0) ? bq : (sel == 1) ? bk : bv;

    const int m    = lane & 15;
    const int half = lane >> 4;
    const int n    = lane & 15;

    const bf16_t* xrow = xb + (size_t)(r0 + m) * Dd;
    const bf16_t* wcol = WT + (size_t)sel * Dd * Dd + (size_t)(c0 + n) * Dd;

    v8f c = {};
    #pragma unroll
    for (int kk = 0; kk < Dd; kk += 32) {
        v8bf alo = *(const v8bf*)(xrow + kk + half * 16);
        v8bf ahi = *(const v8bf*)(xrow + kk + half * 16 + 8);
        v16bf a  = cat16(alo, ahi);
        v16bf b  = *(const v16bf*)(wcol + kk + half * 16);
        c = __builtin_amdgcn_wmma_f32_16x16x32_bf16(false, a, false, b, (short)0, c, false, false);
    }

    const float bi = bias[c0 + n];
    const int h  = (c0 + n) / DHh;
    const int dh = (c0 + n) % DHh;
    bf16_t* qk_base = (sel == 0) ? q_bf : k_bf;   // uniform select
    #pragma unroll
    for (int r = 0; r < 8; ++r) {
        int row = r0 + r + half * 8;              // C layout: M = r (+8 upper lanes)
        int bIdx = row >> 12, nr = row & (Nn - 1);
        float val = c[r] + bi;
        size_t bh = (size_t)bIdx * Hh + h;
        if (sel < 2) qk_base[(bh * Nn + nr) * DHh + dh] = (bf16_t)val;
        else         vT_bf [(bh * DHh + dh) * Nn + nr] = (bf16_t)val;
    }
}

// ---------------- Kernel 2: adjacency-masked flash attention ---------------------
__global__ void attn_kernel(const bf16_t* __restrict__ q_bf,
                            const bf16_t* __restrict__ k_bf,
                            const bf16_t* __restrict__ vT_bf,
                            const unsigned* __restrict__ adjbits,
                            bf16_t* __restrict__ ctx_bf) {
    __shared__ __align__(32) __bf16 lds_p[16 * 64];  // 2 KB prob staging (C->A relayout)

    const int lane = threadIdx.x & 31;
    const int i0   = blockIdx.x * 16;  // query-row tile
    const int h    = blockIdx.y;
    const int b    = blockIdx.z;
    const int bh   = b * Hh + h;
    const int m    = lane & 15;
    const int half = lane >> 4;

    // Q tile (A layout): two contiguous b128 loads
    v16bf aq;
    {
        const bf16_t* qrow = q_bf + ((size_t)bh * Nn + i0 + m) * DHh;
        v8bf lo = *(const v8bf*)(qrow + half * 16);
        v8bf hi = *(const v8bf*)(qrow + half * 16 + 8);
        aq = cat16(lo, hi);
    }

    float mrow[8], lrow[8];
    v8f acc0 = {}, acc1 = {};
    #pragma unroll
    for (int r = 0; r < 8; ++r) { mrow[r] = -3.0e38f; lrow[r] = 0.0f; }

    const float scale = 0.17677669529663687f;   // 1/sqrt(32)
    const bf16_t* kb = k_bf  + (size_t)bh * Nn * DHh;
    const bf16_t* vb = vT_bf + (size_t)bh * DHh * Nn;

    for (int j0 = 0; j0 < Nn; j0 += 64) {
        // 4 K^T tiles as B operands: lane = token column, contiguous per lane
        v16bf bk0 = *(const v16bf*)(kb + (size_t)(j0      + m) * DHh + half * 16);
        v16bf bk1 = *(const v16bf*)(kb + (size_t)(j0 + 16 + m) * DHh + half * 16);
        v16bf bk2 = *(const v16bf*)(kb + (size_t)(j0 + 32 + m) * DHh + half * 16);
        v16bf bk3 = *(const v16bf*)(kb + (size_t)(j0 + 48 + m) * DHh + half * 16);
        v8f s0 = {}, s1 = {}, s2 = {}, s3 = {};
        s0 = __builtin_amdgcn_wmma_f32_16x16x32_bf16(false, aq, false, bk0, (short)0, s0, false, false);
        s1 = __builtin_amdgcn_wmma_f32_16x16x32_bf16(false, aq, false, bk1, (short)0, s1, false, false);
        s2 = __builtin_amdgcn_wmma_f32_16x16x32_bf16(false, aq, false, bk2, (short)0, s2, false, false);
        s3 = __builtin_amdgcn_wmma_f32_16x16x32_bf16(false, aq, false, bk3, (short)0, s3, false, false);

        if (j0 + 64 < Nn) {  // prefetch next K/V chunk (-> global_prefetch_b8)
            __builtin_prefetch(kb + (size_t)(j0 + 64 + m) * DHh, 0, 0);
            __builtin_prefetch(vb + (size_t)m * Nn + j0 + 64, 0, 0);
        }

        float nm[8], corr[8];
        #pragma unroll
        for (int r = 0; r < 8; ++r) {
            int row = i0 + r + half * 8;
            uint2 mw = *(const uint2*)(adjbits + (size_t)row * NW + (j0 >> 5));  // 64 mask bits
            float v0 = ((mw.x >> m)        & 1u) ? s0[r] * scale : -1.0e9f;
            float v1 = ((mw.x >> (16 + m)) & 1u) ? s1[r] * scale : -1.0e9f;
            float v2 = ((mw.y >> m)        & 1u) ? s2[r] * scale : -1.0e9f;
            float v3 = ((mw.y >> (16 + m)) & 1u) ? s3[r] * scale : -1.0e9f;
            s0[r] = v0; s1[r] = v1; s2[r] = v2; s3[r] = v3;
            float mx = fmaxf(fmaxf(v0, v1), fmaxf(v2, v3));
            #pragma unroll
            for (int off = 8; off >= 1; off >>= 1) mx = fmaxf(mx, __shfl_xor(mx, off, 32));
            nm[r]   = fmaxf(mrow[r], mx);
            corr[r] = __expf(mrow[r] - nm[r]);
            mrow[r] = nm[r];
        }
        #pragma unroll
        for (int r = 0; r < 8; ++r) {
            float p0 = __expf(s0[r] - nm[r]);
            float p1 = __expf(s1[r] - nm[r]);
            float p2 = __expf(s2[r] - nm[r]);
            float p3 = __expf(s3[r] - nm[r]);
            float ps = (p0 + p1) + (p2 + p3);
            #pragma unroll
            for (int off = 8; off >= 1; off >>= 1) ps += __shfl_xor(ps, off, 32);
            lrow[r] = lrow[r] * corr[r] + ps;
            int row = r + half * 8;
            lds_p[row * 64 + m]      = (bf16_t)p0;
            lds_p[row * 64 + 16 + m] = (bf16_t)p1;
            lds_p[row * 64 + 32 + m] = (bf16_t)p2;
            lds_p[row * 64 + 48 + m] = (bf16_t)p3;
            acc0[r] *= corr[r];
            acc1[r] *= corr[r];
        }
        asm volatile("s_wait_dscnt 0x0" ::: "memory");  // LDS stores visible before relayout read

        // Probabilities in A layout: ds_load_b128 pairs (j halves 0-31, 32-63)
        v8bf pa = *(const v8bf*)(lds_p + m * 64 + half * 16);
        v8bf pb = *(const v8bf*)(lds_p + m * 64 + half * 16 + 8);
        v8bf pc = *(const v8bf*)(lds_p + m * 64 + 32 + half * 16);
        v8bf pd = *(const v8bf*)(lds_p + m * 64 + 32 + half * 16 + 8);
        v16bf ap0 = cat16(pa, pb);
        v16bf ap1 = cat16(pc, pd);

        // V tiles as B operands (vT: lane = dh column, contiguous over j)
        v16bf bv0 = *(const v16bf*)(vb + (size_t)(m)      * Nn + j0 + half * 16);
        v16bf bv1 = *(const v16bf*)(vb + (size_t)(16 + m) * Nn + j0 + half * 16);
        v16bf bv2 = *(const v16bf*)(vb + (size_t)(m)      * Nn + j0 + 32 + half * 16);
        v16bf bv3 = *(const v16bf*)(vb + (size_t)(16 + m) * Nn + j0 + 32 + half * 16);
        acc0 = __builtin_amdgcn_wmma_f32_16x16x32_bf16(false, ap0, false, bv0, (short)0, acc0, false, false);
        acc1 = __builtin_amdgcn_wmma_f32_16x16x32_bf16(false, ap0, false, bv1, (short)0, acc1, false, false);
        acc0 = __builtin_amdgcn_wmma_f32_16x16x32_bf16(false, ap1, false, bv2, (short)0, acc0, false, false);
        acc1 = __builtin_amdgcn_wmma_f32_16x16x32_bf16(false, ap1, false, bv3, (short)0, acc1, false, false);
    }

    // finalize: ctx = acc / l ; store bf16 [B*N, D] row-major (column = h*32 + dh)
    #pragma unroll
    for (int r = 0; r < 8; ++r) {
        float inv = 1.0f / lrow[r];
        int row = i0 + r + half * 8;
        size_t outbase = ((size_t)b * Nn + row) * Dd + h * DHh;
        ctx_bf[outbase + m]      = (bf16_t)(acc0[r] * inv);
        ctx_bf[outbase + 16 + m] = (bf16_t)(acc1[r] * inv);
    }
}

// ---------------- Kernel 3: output projection + residual -------------------------
__global__ void out_proj_kernel(const bf16_t* __restrict__ ctx_bf,
                                const bf16_t* __restrict__ WT,  // Wo^T at sel 3
                                const float* __restrict__ bo,
                                const float* __restrict__ x,
                                float* __restrict__ out) {
    const int lane = threadIdx.x & 31;
    const int r0   = blockIdx.x * 16;
    const int c0   = blockIdx.y * 16;
    const int m    = lane & 15;
    const int half = lane >> 4;
    const int n    = lane & 15;

    const bf16_t* crow = ctx_bf + (size_t)(r0 + m) * Dd;
    const bf16_t* wcol = WT + (size_t)3 * Dd * Dd + (size_t)(c0 + n) * Dd;

    v8f c = {};
    #pragma unroll
    for (int kk = 0; kk < Dd; kk += 32) {
        v8bf alo = *(const v8bf*)(crow + kk + half * 16);
        v8bf ahi = *(const v8bf*)(crow + kk + half * 16 + 8);
        v16bf a  = cat16(alo, ahi);
        v16bf b  = *(const v16bf*)(wcol + kk + half * 16);
        c = __builtin_amdgcn_wmma_f32_16x16x32_bf16(false, a, false, b, (short)0, c, false, false);
    }

    const float bi = bo[c0 + n];
    #pragma unroll
    for (int r = 0; r < 8; ++r) {
        size_t row = (size_t)r0 + r + half * 8;
        size_t idx = row * Dd + c0 + n;
        out[idx] = x[idx] + c[r] + bi;
    }
}

extern "C" void kernel_launch(void* const* d_in, const int* in_sizes, int n_in,
                              void* d_out, int out_size, void* d_ws, size_t ws_size,
                              hipStream_t stream) {
    const float* x  = (const float*)d_in[0];
    const float* Wq = (const float*)d_in[1];
    const float* bq = (const float*)d_in[2];
    const float* Wk = (const float*)d_in[3];
    const float* bk = (const float*)d_in[4];
    const float* Wv = (const float*)d_in[5];
    const float* bv = (const float*)d_in[6];
    const float* Wo = (const float*)d_in[7];
    const float* bo = (const float*)d_in[8];
    const int*   adj = (const int*)d_in[9];
    float* out = (float*)d_out;

    char* ws = (char*)d_ws;
    const size_t seg = (size_t)Bb * Hh * Nn * DHh * sizeof(unsigned short);  // 2 MB
    bf16_t*   q_bf    = (bf16_t*)(ws);
    bf16_t*   k_bf    = (bf16_t*)(ws + seg);
    bf16_t*   vT_bf   = (bf16_t*)(ws + 2 * seg);
    bf16_t*   ctx_bf  = (bf16_t*)(ws + 3 * seg);
    bf16_t*   xb      = (bf16_t*)(ws + 4 * seg);
    bf16_t*   WT      = (bf16_t*)(ws + 5 * seg);                     // 4 * 32 KB
    unsigned* adjbits = (unsigned*)(ws + 5 * seg + 4 * Dd * Dd * 2); // 2 MB

    // --- prep ---
    const int nx = Bb * Nn * Dd;
    cvt_x_kernel<<<dim3((nx + 255) / 256), dim3(256), 0, stream>>>(x, xb, nx);
    prep_w_kernel<<<dim3((Dd * Dd) / 256, 1, 4), dim3(256), 0, stream>>>(Wq, Wk, Wv, Wo, WT);
    pack_adj_kernel<<<dim3((Nn * NW) / 8), dim3(256), 0, stream>>>(adj, adjbits);

    // --- main pipeline ---
    dim3 g1((Bb * Nn) / 16, Dd / 16, 3);
    qkv_proj_kernel<<<g1, dim3(32, 1, 1), 0, stream>>>(xb, WT, bq, bk, bv, q_bf, k_bf, vT_bf);

    dim3 g2(Nn / 16, Hh, Bb);
    attn_kernel<<<g2, dim3(32, 1, 1), 0, stream>>>(q_bf, k_bf, vT_bf, adjbits, ctx_bf);

    dim3 g3((Bb * Nn) / 16, Dd / 16, 1);
    out_proj_kernel<<<g3, dim3(32, 1, 1), 0, stream>>>(ctx_bf, WT, bo, x, out);
}
